// TransformerBlock_61143154426159
// MI455X (gfx1250) — matrix operand order repaired and checked by reading, not compiled, not run on hardware
//
#include <hip/hip_runtime.h>
#include <math.h>

#define Bc 8
#define Sc 512
#define Ec 1024
#define Hc 16
#define Dc 64

typedef __attribute__((ext_vector_type(16))) __bf16 v16bf;
typedef __attribute__((ext_vector_type(8)))  float  v8f;

// ---------------- bf16 helpers (bit-level, no __bf16 scalar arithmetic) ---------------
__device__ __forceinline__ unsigned short f2bf(float f) {
  union { float f; unsigned u; } c; c.f = f;
  unsigned u = c.u;
  return (unsigned short)((u + 0x7FFFu + ((u >> 16) & 1u)) >> 16);
}
__device__ __forceinline__ float bf2f(unsigned short h) {
  union { unsigned u; float f; } c; c.u = ((unsigned)h) << 16;
  return c.f;
}

// LDS fragment read: 16 bf16 as two 16B ds_load_b128 at off0 and off0+delta.
// A-fragment (16x32, ISA 7.12.2): off0 = (lane>>4)*8,  delta = 16
// B-fragment (32x16):             off0 = (lane>>4)*8,  delta = 16
__device__ __forceinline__ v16bf frag_lds(const unsigned short* row, int off0, int delta) {
  union { uint4 q[2]; v16bf v; } t;
  t.q[0] = *(const uint4*)(row + off0);
  t.q[1] = *(const uint4*)(row + off0 + delta);
  return t.v;
}

__device__ __forceinline__ float wred_max(float v) {
  #pragma unroll
  for (int o = 16; o; o >>= 1) v = fmaxf(v, __shfl_xor(v, o, 32));
  return v;
}
__device__ __forceinline__ float wred_sum(float v) {
  #pragma unroll
  for (int o = 16; o; o >>= 1) v += __shfl_xor(v, o, 32);
  return v;
}

// =====================================================================================
// Generic batched bf16 WMMA GEMM:  C[m,n] = act( sum_k A[m,k]*Bw[n,k] (+A2@B2) + bias + preadd )
// Block: 256 thr = 8 waves; block tile 128x128; wave tile 32x64 (2x4 wmma 16x16 tiles).
// Data path: double-buffered async global->LDS tiles (ASYNCcnt), ds_load_b128 fragments.
// flags: bit0 = bf16 output, bit2 = Transformer-XL rel-shift scatter-add into f32 C.
// =====================================================================================
__global__ __launch_bounds__(256)
void gtrxl_gemm_wmma_bf16(const unsigned short* __restrict__ A,
                          const unsigned short* __restrict__ Bw,
                          const unsigned short* __restrict__ A2,
                          const unsigned short* __restrict__ B2,
                          const float* __restrict__ bias,
                          void* __restrict__ Cout,
                          int M, int N, int K, int K2,
                          int lda, int ldb, int ldc,
                          long long sAb, long long sAh,
                          long long sBb, long long sBh,
                          long long sCb, long long sCh,
                          int nh, float preadd, int act, int flags) {
  // [buf][A=0/B=1][row][K padded 32->40] : 40960 B, bank-conflict-free fragment reads
  __shared__ unsigned short smem[2][2][128][40];

  const int tid  = threadIdx.x;
  const int lane = tid & 31;
  const int wave = tid >> 5;
  const int wm = wave >> 1;        // 0..3
  const int wn = wave & 1;         // 0..1

  const int bz = blockIdx.z;
  const int b  = bz / nh;
  const int h  = bz - b * nh;
  const unsigned short* Ab = A  + (size_t)b * sAb + (size_t)h * sAh;
  const unsigned short* Bb = Bw + (size_t)b * sBb + (size_t)h * sBh;
  const long long cbase = (long long)b * sCb + (long long)h * sCh;

  const int mblk = blockIdx.y * 128;
  const int nblk = blockIdx.x * 128;
  const int m0 = mblk + wm * 32;
  const int n0 = nblk + wn * 64;

  const int lr   = lane & 15;
  const int kh   = (lane >> 4) * 8;    // A half-wave K offset
  const int kh16 = (lane >> 4) * 16;   // unused (B fragments use kh as well)

  v8f c[2][4];
  #pragma unroll
  for (int i = 0; i < 2; ++i)
    #pragma unroll
    for (int j = 0; j < 4; ++j)
      c[i][j] = (v8f){0.f,0.f,0.f,0.f,0.f,0.f,0.f,0.f};

  const unsigned short* SAs[2] = {Ab, A2};
  const unsigned short* SBs[2] = {Bb, B2};
  const int KKs[2] = {K, K2};
  const int nsrc = (A2 != nullptr) ? 2 : 1;

  // Cooperative async fill of one 128x32 bf16 tile into LDS (512 x 16B chunks,
  // 2 per thread) via gfx1250 GLOBAL_LOAD_ASYNC_TO_LDS_B128 (ASYNCcnt-tracked).
  auto fill = [&](const unsigned short* base, int ld, int lim, int rowbase, int kb,
                  int bufi, int ab) {
    unsigned lbase = (unsigned)(size_t)(&smem[bufi][ab][0][0]);  // flat[31:0] == LDS offset
    #pragma unroll
    for (int t = 0; t < 2; ++t) {
      int cch = tid + t * 256;
      int r = cch >> 2, seg = cch & 3;          // tile row, 16B segment within 64B row
      int gr = rowbase + r; if (gr >= lim) gr = lim - 1;
      const unsigned short* g = base + (size_t)gr * (size_t)ld + (size_t)(kb + seg * 8);
      unsigned l = lbase + (unsigned)(r * 80 + seg * 16);
      asm volatile("global_load_async_to_lds_b128 %0, %1, off"
                   :: "v"(l), "v"(g) : "memory");
    }
  };

  int buf = 0;
  for (int s = 0; s < nsrc; ++s) {
    const unsigned short* SA = SAs[s];
    const unsigned short* SB = SBs[s];
    const int Ks = KKs[s];

    // prime first buffer
    fill(SA, lda, M, mblk, 0, buf, 0);
    fill(SB, ldb, N, nblk, 0, buf, 1);
    asm volatile("s_wait_asynccnt 0" ::: "memory");
    __syncthreads();

    for (int kb = 0; kb < Ks; kb += 32) {
      const bool more = (kb + 32 < Ks);
      if (more) {                      // overlap: fill next buffer while computing
        fill(SA, lda, M, mblk, kb + 32, buf ^ 1, 0);
        fill(SB, ldb, N, nblk, kb + 32, buf ^ 1, 1);
      }
      v16bf af[2], bfr[4];
      #pragma unroll
      for (int i = 0; i < 2; ++i)
        af[i] = frag_lds(&smem[buf][0][wm * 32 + i * 16 + lr][0], kh, 16);
      #pragma unroll
      for (int j = 0; j < 4; ++j)
        bfr[j] = frag_lds(&smem[buf][1][wn * 64 + j * 16 + lr][0], kh, 16);
      #pragma unroll
      for (int i = 0; i < 2; ++i)
        #pragma unroll
        for (int j = 0; j < 4; ++j)
          c[i][j] = __builtin_amdgcn_wmma_f32_16x16x32_bf16(
              false, af[i], false, bfr[j], (short)0, c[i][j], false, false);
      asm volatile("s_wait_asynccnt 0" ::: "memory");
      __syncthreads();
      buf ^= 1;
    }
  }

  // ------------------------- epilogue -------------------------
  const int rhalf = (lane >> 4) * 8;
  #pragma unroll
  for (int i = 0; i < 2; ++i) {
    #pragma unroll
    for (int j = 0; j < 4; ++j) {
      const int col = n0 + j * 16 + lr;
      #pragma unroll
      for (int e = 0; e < 8; ++e) {
        const int row = m0 + i * 16 + rhalf + e;   // C layout: VGPR e -> M = e + 8*half
        if (row >= M || col >= N) continue;
        float v = c[i][j][e];
        if (bias) v += bias[col];
        v += preadd;
        if      (act == 1) v = 1.f / (1.f + __expf(-v));   // sigmoid
        else if (act == 2) v = tanhf(v);
        else if (act == 3) v = fmaxf(v, 0.f);              // relu
        if (flags & 4) {
          // rel-shift scatter-add (bijective): n = row*(S+1)+col+1 ; i=n/S-1 ; j=n%S
          long long nn = (long long)row * (ldc + 1) + col + 1;
          int ii = (int)(nn / ldc) - 1;
          int jj = (int)(nn % ldc);
          if (ii >= 0 && ii < M)
            ((float*)Cout)[cbase + (long long)ii * ldc + jj] += v;
        } else if (flags & 1) {
          ((unsigned short*)Cout)[cbase + (long long)row * ldc + col] = f2bf(v);
        } else {
          ((float*)Cout)[cbase + (long long)row * ldc + col] = v;
        }
      }
    }
  }
}

// ================================ LayerNorm (f32 -> bf16) ============================
__global__ __launch_bounds__(256)
void gtrxl_layernorm(const float* __restrict__ x, const float* __restrict__ g,
                     const float* __restrict__ bta, unsigned short* __restrict__ out) {
  const int row = blockIdx.x;
  const float* p = x + (size_t)row * Ec;
  float s = 0.f, s2 = 0.f;
  for (int i = threadIdx.x; i < Ec; i += 256) { float v = p[i]; s += v; s2 += v * v; }
  __shared__ float sh[16];
  s = wred_sum(s); s2 = wred_sum(s2);
  const int lane = threadIdx.x & 31, w = threadIdx.x >> 5;
  if (!lane) { sh[w] = s; sh[8 + w] = s2; }
  __syncthreads();
  float a = 0.f, a2 = 0.f;
  #pragma unroll
  for (int i = 0; i < 8; ++i) { a += sh[i]; a2 += sh[8 + i]; }
  const float mean = a * (1.f / Ec);
  const float var  = a2 * (1.f / Ec) - mean * mean;
  const float inv  = rsqrtf(var + 1e-5f);
  for (int i = threadIdx.x; i < Ec; i += 256)
    out[(size_t)row * Ec + i] = f2bf((p[i] - mean) * inv * g[i] + bta[i]);
}

// ========================= masked scaled softmax (f32 -> bf16) =======================
__global__ __launch_bounds__(256)
void gtrxl_softmax(const float* __restrict__ sc, unsigned short* __restrict__ attn) {
  const int row = blockIdx.x;                 // b*H*S + h*S + s
  const int s   = row & (Sc - 1);
  const float* p = sc + (size_t)row * Sc;
  unsigned short* o = attn + (size_t)row * Sc;
  float vals[2];
  float mx = -3.4e38f;
  #pragma unroll
  for (int it = 0; it < 2; ++it) {
    int t = threadIdx.x + it * 256;
    float v = (t <= s) ? (p[t] * 0.125f) : -1e9f;   // 1/sqrt(64), causal mask
    vals[it] = v;
    mx = fmaxf(mx, v);
  }
  __shared__ float shm[8], shs[8];
  mx = wred_max(mx);
  const int lane = threadIdx.x & 31, w = threadIdx.x >> 5;
  if (!lane) shm[w] = mx;
  __syncthreads();
  mx = shm[0];
  #pragma unroll
  for (int i = 1; i < 8; ++i) mx = fmaxf(mx, shm[i]);
  float sum = 0.f;
  #pragma unroll
  for (int it = 0; it < 2; ++it) { vals[it] = __expf(vals[it] - mx); sum += vals[it]; }
  sum = wred_sum(sum);
  if (!lane) shs[w] = sum;
  __syncthreads();
  sum = 0.f;
  #pragma unroll
  for (int i = 0; i < 8; ++i) sum += shs[i];
  const float inv = 1.f / sum;
  #pragma unroll
  for (int it = 0; it < 2; ++it) o[threadIdx.x + it * 256] = f2bf(vals[it] * inv);
}

// ================================ elementwise kernels ================================
__global__ void gtrxl_cvt_bf16(const float* __restrict__ in, unsigned short* __restrict__ out, int n) {
  int i = blockIdx.x * 256 + threadIdx.x;
  if (i < n) out[i] = f2bf(in[i]);
}
__global__ void gtrxl_add_headbias(const float* __restrict__ q, const float* __restrict__ ub,
                                   unsigned short* __restrict__ out, int n) {
  int i = blockIdx.x * 256 + threadIdx.x;
  if (i < n) out[i] = f2bf(q[i] + ub[i & (Ec - 1)]);
}
__global__ void gtrxl_posenc(unsigned short* __restrict__ pe) {
  int idx = blockIdx.x * 256 + threadIdx.x;
  if (idx >= Sc * Ec) return;
  int s = idx >> 10, e = idx & (Ec - 1);
  float pos = (float)(Sc - 1 - s);
  int half = Ec >> 1;
  int i = (e < half) ? e : e - half;
  float inv = __expf(-((float)(2 * i) / (float)Ec) * 9.210340371976184f); // ln(10000)
  float ang = pos * inv;
  pe[idx] = f2bf((e < half) ? __sinf(ang) : __cosf(ang));
}
__global__ void gtrxl_transpose_v(const unsigned short* __restrict__ v, unsigned short* __restrict__ vT) {
  int idx = blockIdx.x * 256 + threadIdx.x;
  if (idx >= Bc * Sc * Ec) return;
  int d = idx & 63, h = (idx >> 6) & 15, s = (idx >> 10) & 511, b = idx >> 19;
  vT[(((size_t)(b * Hc + h) * Dc + d) * Sc) + s] = v[idx];   // [B,S,H,D] -> [B,H,D,S]
}
__global__ void gtrxl_rx(const unsigned short* __restrict__ r, const float* __restrict__ x,
                         unsigned short* __restrict__ out, int n) {
  int i = blockIdx.x * 256 + threadIdx.x;
  if (i < n) out[i] = f2bf(bf2f(r[i]) * x[i]);
}
__global__ void gtrxl_gru_combine(const float* __restrict__ z, const float* __restrict__ x,
                                  const float* __restrict__ hpre, float* __restrict__ out, int n) {
  int i = blockIdx.x * 256 + threadIdx.x;
  if (i < n) {
    float zz = z[i];
    out[i] = (1.f - zz) * x[i] + zz * tanhf(hpre[i]);
  }
}

// =====================================================================================
extern "C" void kernel_launch(void* const* d_in, const int* in_sizes, int n_in,
                              void* d_out, int out_size, void* d_ws, size_t ws_size,
                              hipStream_t stream) {
  (void)in_sizes; (void)n_in; (void)out_size; (void)ws_size;
  const float* query = (const float*)d_in[0];
  const float* key   = (const float*)d_in[1];
  const float* ln1_g = (const float*)d_in[4];
  const float* ln1_b = (const float*)d_in[5];
  const float* ln2_g = (const float*)d_in[6];
  const float* ln2_b = (const float*)d_in[7];
  const float* bq = (const float*)d_in[9];
  const float* bk = (const float*)d_in[11];
  const float* bv = (const float*)d_in[13];
  const float* bo = (const float*)d_in[16];
  const float* u_bias = (const float*)d_in[17];
  const float* v_bias = (const float*)d_in[18];
  const float* fc_b   = (const float*)d_in[32];

  const size_t MSE  = (size_t)Bc * Sc * Ec;        // 4194304
  const size_t EE   = (size_t)Ec * Ec;             // 1048576
  const size_t SE   = (size_t)Sc * Ec;             // 524288
  const size_t BHSS = (size_t)Bc * Hc * Sc * Sc;   // 33554432

  char* ws = (char*)d_ws;
  size_t off = 0;
  auto alloc = [&](size_t bytes) -> char* {
    char* p = ws + off;
    off += (bytes + 255) & ~(size_t)255;
    return p;
  };
  unsigned short* WBF  = (unsigned short*)alloc(18 * EE * 2);  // 18 bf16 weight matrices
  unsigned short* NQ   = (unsigned short*)alloc(MSE * 2);
  unsigned short* NK   = (unsigned short*)alloc(MSE * 2);
  unsigned short* PE   = (unsigned short*)alloc(SE * 2);
  float*          QF   = (float*)         alloc(MSE * 4);
  unsigned short* QU   = (unsigned short*)alloc(MSE * 2);
  unsigned short* QV   = (unsigned short*)alloc(MSE * 2);
  unsigned short* KB   = (unsigned short*)alloc(MSE * 2);
  unsigned short* VB   = (unsigned short*)alloc(MSE * 2);
  unsigned short* VT   = (unsigned short*)alloc(MSE * 2);
  unsigned short* RB   = (unsigned short*)alloc(SE * 2);
  float*          SC   = (float*)         alloc(BHSS * 4);
  unsigned short* AT   = (unsigned short*)alloc(BHSS * 2);
  unsigned short* AV   = (unsigned short*)alloc(MSE * 2);
  unsigned short* YB   = (unsigned short*)alloc(MSE * 2);
  unsigned short* XB   = (unsigned short*)alloc(MSE * 2);
  unsigned short* R1   = (unsigned short*)alloc(MSE * 2);
  float*          Z1   = (float*)         alloc(MSE * 4);
  unsigned short* RX1  = (unsigned short*)alloc(MSE * 2);
  float*          H1   = (float*)         alloc(MSE * 4);
  float*          O1   = (float*)         alloc(MSE * 4);
  unsigned short* O1B  = (unsigned short*)alloc(MSE * 2);
  unsigned short* LN2B = (unsigned short*)alloc(MSE * 2);
  unsigned short* EF   = (unsigned short*)alloc(MSE * 2);
  unsigned short* R2   = (unsigned short*)alloc(MSE * 2);
  float*          Z2   = (float*)         alloc(MSE * 4);
  unsigned short* RX2  = (unsigned short*)alloc(MSE * 2);
  float*          H2   = (float*)         alloc(MSE * 4);

  const int ewB = (int)((MSE + 255) / 256);

  // ---- convert 18 weight matrices f32 -> bf16 ----
  const int widx[18] = {8,10,12,14,15, 19,20,21,22,23,24, 25,26,27,28,29,30, 31};
  for (int i = 0; i < 18; ++i)
    gtrxl_cvt_bf16<<<(int)((EE + 255) / 256), 256, 0, stream>>>(
        (const float*)d_in[widx[i]], WBF + (size_t)i * EE, (int)EE);
  unsigned short* Wq = WBF;            unsigned short* Wk = WBF + 1 * EE;
  unsigned short* Wv = WBF + 2 * EE;   unsigned short* Wp = WBF + 3 * EE;
  unsigned short* Wo = WBF + 4 * EE;
  unsigned short* G1[6]; for (int i = 0; i < 6; ++i) G1[i] = WBF + (size_t)(5  + i) * EE;
  unsigned short* G2[6]; for (int i = 0; i < 6; ++i) G2[i] = WBF + (size_t)(11 + i) * EE;
  unsigned short* FW = WBF + 17 * EE;
  // order: [0]=Wr [1]=Ur [2]=Wz [3]=Uz [4]=Wg [5]=Ug

  auto gemm = [&](const unsigned short* A, const unsigned short* Bw,
                  const unsigned short* A2, const unsigned short* B2,
                  const float* bias, void* C,
                  int M, int N, int K, int K2, int lda, int ldb, int ldc,
                  long long sAb, long long sAh, long long sBb, long long sBh,
                  long long sCb, long long sCh, int nb, int nh,
                  float preadd, int act, int flags) {
    dim3 g((N + 127) / 128, (M + 127) / 128, nb * nh);
    gtrxl_gemm_wmma_bf16<<<g, dim3(256), 0, stream>>>(
        A, Bw, A2, B2, bias, C, M, N, K, K2, lda, ldb, ldc,
        sAb, sAh, sBb, sBh, sCb, sCh, nh, preadd, act, flags);
  };

  // ---- LayerNorm1 on query/key; positional encoding; x := bf16(query) ----
  gtrxl_layernorm<<<Bc * Sc, 256, 0, stream>>>(query, ln1_g, ln1_b, NQ);
  gtrxl_layernorm<<<Bc * Sc, 256, 0, stream>>>(key,   ln1_g, ln1_b, NK);
  gtrxl_posenc<<<(int)((SE + 255) / 256), 256, 0, stream>>>(PE);
  gtrxl_cvt_bf16<<<ewB, 256, 0, stream>>>(query, XB, (int)MSE);

  // ---- projections ----
  gemm(NQ, Wq, 0, 0, bq, QF, 4096, 1024, 1024, 0, 1024, 1024, 1024, 0,0,0,0,0,0, 1,1, 0.f, 0, 0);
  gtrxl_add_headbias<<<ewB, 256, 0, stream>>>(QF, u_bias, QU, (int)MSE);   // q + u
  gtrxl_add_headbias<<<ewB, 256, 0, stream>>>(QF, v_bias, QV, (int)MSE);   // q + v
  gemm(NK, Wk, 0, 0, bk, KB, 4096, 1024, 1024, 0, 1024, 1024, 1024, 0,0,0,0,0,0, 1,1, 0.f, 0, 1);
  gemm(NK, Wv, 0, 0, bv, VB, 4096, 1024, 1024, 0, 1024, 1024, 1024, 0,0,0,0,0,0, 1,1, 0.f, 0, 1);
  gtrxl_transpose_v<<<ewB, 256, 0, stream>>>(VB, VT);                      // [B,H,D,S]
  gemm(PE, Wp, 0, 0, 0, RB, 512, 1024, 1024, 0, 1024, 1024, 1024, 0,0,0,0,0,0, 1,1, 0.f, 0, 1);

  // ---- attention scores: AC then BD (rel-shift scatter-add), batched over (b,h) ----
  gemm(QU, KB, 0, 0, 0, SC, 512, 512, 64, 0, 1024, 1024, 512,
       (long long)Sc*Ec, Dc, (long long)Sc*Ec, Dc, (long long)Hc*Sc*Sc, (long long)Sc*Sc,
       Bc, Hc, 0.f, 0, 0);
  gemm(QV, RB, 0, 0, 0, SC, 512, 512, 64, 0, 1024, 1024, 512,
       (long long)Sc*Ec, Dc, 0, Dc, (long long)Hc*Sc*Sc, (long long)Sc*Sc,
       Bc, Hc, 0.f, 0, 4);
  gtrxl_softmax<<<Bc * Hc * Sc, 256, 0, stream>>>(SC, AT);

  // ---- attn @ V  ([B,H]: [S,S] x [S,D]) then Wo ----
  gemm(AT, VT, 0, 0, 0, AV, 512, 64, 512, 0, 512, 512, 1024,
       (long long)Hc*Sc*Sc, (long long)Sc*Sc, (long long)Hc*Dc*Sc, (long long)Dc*Sc,
       (long long)Sc*Ec, Dc, Bc, Hc, 0.f, 0, 1);
  gemm(AV, Wo, 0, 0, bo, YB, 4096, 1024, 1024, 0, 1024, 1024, 1024, 0,0,0,0,0,0, 1,1, 0.f, 0, 1);

  // ---- GRU gate 1 (x=query, y=Y): fused dual-K GEMMs ----
  gemm(YB, G1[0], XB, G1[1], 0, R1, 4096, 1024, 1024, 1024, 1024, 1024, 1024,
       0,0,0,0,0,0, 1,1, 0.f,  1, 1);                                  // r = sigmoid
  gemm(YB, G1[2], XB, G1[3], 0, Z1, 4096, 1024, 1024, 1024, 1024, 1024, 1024,
       0,0,0,0,0,0, 1,1, -2.0f, 1, 0);                                 // z = sigmoid(.. - 2)
  gtrxl_rx<<<ewB, 256, 0, stream>>>(R1, query, RX1, (int)MSE);         // r*x
  gemm(YB, G1[4], RX1, G1[5], 0, H1, 4096, 1024, 1024, 1024, 1024, 1024, 1024,
       0,0,0,0,0,0, 1,1, 0.f,  0, 0);                                  // h_pre
  gtrxl_gru_combine<<<ewB, 256, 0, stream>>>(Z1, query, H1, O1, (int)MSE);

  // ---- LN2 + FC(relu) ----
  gtrxl_layernorm<<<Bc * Sc, 256, 0, stream>>>(O1, ln2_g, ln2_b, LN2B);
  gemm(LN2B, FW, 0, 0, fc_b, EF, 4096, 1024, 1024, 0, 1024, 1024, 1024,
       0,0,0,0,0,0, 1,1, 0.f, 3, 1);                                   // relu -> bf16
  gtrxl_cvt_bf16<<<ewB, 256, 0, stream>>>(O1, O1B, (int)MSE);

  // ---- GRU gate 2 (x=out1, y=Ef) ----
  gemm(EF, G2[0], O1B, G2[1], 0, R2, 4096, 1024, 1024, 1024, 1024, 1024, 1024,
       0,0,0,0,0,0, 1,1, 0.f,  1, 1);
  gemm(EF, G2[2], O1B, G2[3], 0, Z2, 4096, 1024, 1024, 1024, 1024, 1024, 1024,
       0,0,0,0,0,0, 1,1, -2.0f, 1, 0);
  gtrxl_rx<<<ewB, 256, 0, stream>>>(R2, O1, RX2, (int)MSE);
  gemm(EF, G2[4], RX2, G2[5], 0, H2, 4096, 1024, 1024, 1024, 1024, 1024, 1024,
       0,0,0,0,0,0, 1,1, 0.f,  0, 0);
  gtrxl_gru_combine<<<ewB, 256, 0, stream>>>(Z2, O1, H2, (float*)d_out, (int)MSE);
}
